// SlidingWindowAttention_58849641890634
// MI455X (gfx1250) — matrix operand order repaired and checked by reading, not compiled, not run on hardware
//
#include <hip/hip_runtime.h>
#include <hip/hip_bf16.h>

// ---------------------------------------------------------------------------
// Sliding-window attention for MI455X (gfx1250).
// All GEMM work on v_wmma_f32_16x16x32_f16; GEMM tiles staged via the
// Tensor Data Mover (tensor_load_to_lds, TENSORcnt double-buffering).
// B=2, T=4096, D=1024, NH=16, H=64, CW=128, NB=32.
// ---------------------------------------------------------------------------

typedef __attribute__((ext_vector_type(16))) _Float16     v16h;
typedef __attribute__((ext_vector_type(8)))  _Float16     v8h;
typedef __attribute__((ext_vector_type(4)))  _Float16     v4h;
typedef __attribute__((ext_vector_type(8)))  float        v8f;
typedef __attribute__((ext_vector_type(4)))  unsigned int v4u;
typedef __attribute__((ext_vector_type(8)))  int          v8i;
typedef __attribute__((ext_vector_type(4)))  int          v4i;

#define WMMA_F16(a, b, c) \
  __builtin_amdgcn_wmma_f32_16x16x32_f16(false, (a), false, (b), (short)0, (c), false, false)

static constexpr int Bc  = 2;
static constexpr int Tt  = 4096;
static constexpr int Dd  = 1024;
static constexpr int NHh = 16;
static constexpr int Hd  = 64;
static constexpr int CWw = 128;
static constexpr int NBb = 32;
static constexpr int Mrows = Bc * Tt;   // 8192

// ---------------------------------------------------------------------------
// TDM: DMA a 2D f16 tile (tile_d0 contiguous elems x tile_d1 rows, row stride
// stride_elems) from global memory into LDS at lds_byte_off.
// D# layout per CDNA5 ISA (§8.3/8.4); tensor_dim == tile_dim
// so no OOB clipping ever triggers. 6-arg builtin form (clang-23 headers):
// (g0, g1, g2, g3, g_extra, cpol); groups 2/3/extra zero for a 2-D tensor.
// ---------------------------------------------------------------------------
__device__ inline void tdm_load_tile_f16(unsigned lds_byte_off, const _Float16* gptr,
                                         unsigned tile_d0, unsigned tile_d1,
                                         unsigned long long stride_elems) {
  const unsigned long long ga = (unsigned long long)gptr;
  v4u g0;
  g0[0] = 1u;                                                  // count=1, user desc
  g0[1] = lds_byte_off;                                        // lds_addr
  g0[2] = (unsigned)ga;                                        // global_addr[31:0]
  g0[3] = (unsigned)((ga >> 32) & 0x01FFFFFFu) | (2u << 30);   // addr[56:32] | type=2
  v8i g1;
  g1[0] = (int)(1u << 16);                                     // data_size=1 (2 bytes)
  g1[1] = (int)((tile_d0 & 0xFFFFu) << 16);                    // tensor_dim0[15:0]
  g1[2] = (int)(((tile_d0 >> 16) & 0xFFFFu) |                  // tensor_dim0[31:16]
                ((tile_d1 & 0xFFFFu) << 16));                  // tensor_dim1[15:0]
  g1[3] = (int)(((tile_d1 >> 16) & 0xFFFFu) |                  // tensor_dim1[31:16]
                ((tile_d0 & 0xFFFFu) << 16));                  // tile_dim0
  g1[4] = (int)(tile_d1 & 0xFFFFu);                            // tile_dim1 (tile_dim2=0)
  g1[5] = (int)(unsigned)(stride_elems & 0xFFFFFFFFull);       // dim0_stride[31:0]
  g1[6] = (int)(unsigned)((stride_elems >> 32) & 0xFFFFull);   // dim0_stride[47:32]
  g1[7] = 0;                                                   // dim1_stride unused
  const v4i z4 = {0, 0, 0, 0};
  const v8i z8 = {0, 0, 0, 0, 0, 0, 0, 0};
  __builtin_amdgcn_tensor_load_to_lds(g0, g1, z4, z4, z8, 0);
}

__device__ inline unsigned lds_off(const void* p) {
  return (unsigned)(unsigned long long)p;   // flat addr low 32 bits = LDS offset
}

// ---------------------------------------------------------------------------
// Fragment loaders (LDS -> VGPR), matching CDNA5 16-bit WMMA layouts.
// A (16x32): lane={M=lane%16, half}; chunks at K=8*half and K=16+8*half.
// B (32x16) stored N-major/K-contiguous: lane={N=lane%16, half}; chunks at K=8*half and K=16+8*half.
// ---------------------------------------------------------------------------
__device__ inline v16h load_a_frag(const _Float16* base, int ld) {
  const int lane = threadIdx.x & 31;
  const int row  = lane & 15;
  const int half = lane >> 4;
  const _Float16* p = base + row * ld + 8 * half;
  v8h lo = *(const v8h*)(p);
  v8h hi = *(const v8h*)(p + 16);
  v16h r;
#pragma unroll
  for (int i = 0; i < 8; ++i) { r[i] = lo[i]; r[i + 8] = hi[i]; }
  return r;
}

__device__ inline v16h load_b_frag(const _Float16* base, int ld) {
  const int lane = threadIdx.x & 31;
  const int n    = lane & 15;
  const int half = lane >> 4;
  const _Float16* p = base + n * ld + 8 * half;
  v8h lo = *(const v8h*)(p);
  v8h hi = *(const v8h*)(p + 16);
  v16h r;
#pragma unroll
  for (int i = 0; i < 8; ++i) { r[i] = lo[i]; r[i + 8] = hi[i]; }
  return r;
}

// ---------------------------------------------------------------------------
// fp32 -> f16 conversion (vectorized x4)
// ---------------------------------------------------------------------------
__global__ __launch_bounds__(256)
void f32_to_f16_kernel(const float* __restrict__ src, _Float16* __restrict__ dst, int n) {
  int i = (blockIdx.x * 256 + threadIdx.x) * 4;
  if (i + 3 < n) {
    float4 v = *(const float4*)(src + i);
    v4h o;
    o[0] = (_Float16)v.x; o[1] = (_Float16)v.y;
    o[2] = (_Float16)v.z; o[3] = (_Float16)v.w;
    *(v4h*)(dst + i) = o;
  } else {
    for (; i < n; ++i) dst[i] = (_Float16)src[i];
  }
}

// ---------------------------------------------------------------------------
// fp32 [K][N] -> f16 [N][K] transposing conversion (LDS-tiled, coalesced).
// Makes weight B-tiles K-contiguous so the TDM can stage them directly.
// ---------------------------------------------------------------------------
__global__ __launch_bounds__(256)
void f32_to_f16_transpose_kernel(const float* __restrict__ src, _Float16* __restrict__ dst,
                                 int Kdim, int Ndim) {
  __shared__ float tile[32][33];
  const int k0 = blockIdx.y * 32;
  const int n0 = blockIdx.x * 32;
  const int tx = threadIdx.x;        // 0..31
  const int ty = threadIdx.y;        // 0..7
#pragma unroll
  for (int r = ty; r < 32; r += 8)
    tile[r][tx] = src[(long)(k0 + r) * Ndim + n0 + tx];
  __syncthreads();
#pragma unroll
  for (int r = ty; r < 32; r += 8)
    dst[(long)(n0 + r) * Kdim + k0 + tx] = (_Float16)tile[tx][r];
}

// ---------------------------------------------------------------------------
// 128x128-tile GEMM, C = A[MxK] * Wt[NxK]^T (+bias). 256 threads = 8 wave32s,
// each wave owns a 32x64 tile = 2x4 WMMA accumulators.
// Tiles staged with TDM (tensor_load_to_lds), double-buffered on TENSORcnt:
// wave 0 issues the next A/B pair, waits tensorcnt<=2 (current pair done,
// next pair in flight), barrier releases the WMMA waves.
// ---------------------------------------------------------------------------
template <bool F16OUT>
__global__ __launch_bounds__(256)
void gemm128_wmma_tdm(const _Float16* __restrict__ A, const _Float16* __restrict__ Wt,
                      _Float16* __restrict__ outH, float* __restrict__ outF,
                      const float* __restrict__ bias, int M, int N, int K) {
  __shared__ _Float16 sA[2][128 * 32];   // [m][k] tiles, ping-pong
  __shared__ _Float16 sB[2][128 * 32];   // [n][k] tiles, ping-pong

  const int tid  = threadIdx.x;
  const int lane = tid & 31;
  const int wid  = tid >> 5;
  const int wm   = wid >> 1;          // 0..3
  const int wn   = wid & 1;           // 0..1
  const long m0  = (long)blockIdx.y * 128;
  const long n0  = (long)blockIdx.x * 128;

  const _Float16* Abase = A  + m0 * K;
  const _Float16* Bbase = Wt + n0 * K;

  v8f acc[2][4];
#pragma unroll
  for (int i = 0; i < 2; ++i)
#pragma unroll
    for (int j = 0; j < 4; ++j)
#pragma unroll
      for (int v = 0; v < 8; ++v) acc[i][j][v] = 0.0f;

  const int nIter = K >> 5;           // K/32

  if (wid == 0) {                     // prologue: DMA first tile pair into buf 0
    tdm_load_tile_f16(lds_off(sA[0]), Abase, 32, 128, (unsigned long long)K);
    tdm_load_tile_f16(lds_off(sB[0]), Bbase, 32, 128, (unsigned long long)K);
  }

  for (int it = 0; it < nIter; ++it) {
    const int cur = it & 1;
    if (wid == 0) {
      if (it + 1 < nIter) {           // prefetch next pair into the other buffer
        const int k1 = (it + 1) << 5;
        tdm_load_tile_f16(lds_off(sA[1 - cur]), Abase + k1, 32, 128, (unsigned long long)K);
        tdm_load_tile_f16(lds_off(sB[1 - cur]), Bbase + k1, 32, 128, (unsigned long long)K);
        __builtin_amdgcn_s_wait_tensorcnt(2);   // current pair complete
      } else {
        __builtin_amdgcn_s_wait_tensorcnt(0);   // drain
      }
    }
    __syncthreads();

    v16h af[2], bf[4];
#pragma unroll
    for (int i = 0; i < 2; ++i) af[i] = load_a_frag(sA[cur] + (wm * 32 + i * 16) * 32, 32);
#pragma unroll
    for (int j = 0; j < 4; ++j) bf[j] = load_b_frag(sB[cur] + (wn * 64 + j * 16) * 32, 32);
#pragma unroll
    for (int i = 0; i < 2; ++i)
#pragma unroll
      for (int j = 0; j < 4; ++j) acc[i][j] = WMMA_F16(af[i], bf[j], acc[i][j]);

    __syncthreads();                  // reads done before TDM overwrites this buffer
  }

  // epilogue: C layout lane={N=lane%16, half=lane/16}, VGPR v -> M = v + 8*half
  const int cn = lane & 15;
  const int hf = lane >> 4;
#pragma unroll
  for (int i = 0; i < 2; ++i)
#pragma unroll
    for (int j = 0; j < 4; ++j)
#pragma unroll
      for (int v = 0; v < 8; ++v) {
        const long row = m0 + wm * 32 + i * 16 + v + 8 * hf;
        const long col = n0 + wn * 64 + j * 16 + cn;
        const float val = acc[i][j][v];
        if constexpr (F16OUT) {
          outH[row * N + col] = (_Float16)val;
        } else {
          outF[row * N + col] = val + bias[col];
        }
      }
}

// ---------------------------------------------------------------------------
// Sliding-window attention. One block per (b, query-block, head).
// S = Q K^T / 8 (masked), P = softmax(S), O = P V. All matmuls on WMMA.
// LDS (dynamic, 80KB): Vt[64][256] | Q[128][64] | K[256][64];
// P[128][128] (two column-halves) overlays the dead Q/K region.
// ---------------------------------------------------------------------------
__global__ __launch_bounds__(256)
void sliding_attn_wmma(const _Float16* __restrict__ Qh, const _Float16* __restrict__ Kh,
                       const _Float16* __restrict__ Vh, _Float16* __restrict__ AOh) {
  extern __shared__ char smem[];
  _Float16* sVt = (_Float16*)(smem);            // [64][256]  32KB
  _Float16* sQ  = (_Float16*)(smem + 32768);    // [128][64]  16KB
  _Float16* sK  = (_Float16*)(smem + 49152);    // [256][64]  32KB
  _Float16* sP  = sQ;                           // [128][128] 32KB overlay

  const int tid  = threadIdx.x;
  const int lane = tid & 31;
  const int w    = tid >> 5;                    // wave id: query rows w*16..w*16+15
  const int h    = blockIdx.x % NHh;
  const int nb   = (blockIdx.x / NHh) % NBb;
  const int b    = blockIdx.x / (NHh * NBb);

  // ---- stage Q (2 threads per row) ----
  {
    const int r     = tid >> 1;
    const int cpart = (tid & 1) * 32;
    const _Float16* g = Qh + ((long)(b * Tt + nb * CWw + r)) * Dd + h * Hd + cpart;
#pragma unroll
    for (int c = 0; c < 32; c += 8)
      *(v8h*)(sQ + r * 64 + cpart + c) = *(const v8h*)(g + c);
  }
  // ---- stage K slab [256][64] and V slab transposed [64][256] ----
  {
    const int j    = tid;                       // 0..255 key-slab row
    const int krow = (nb - 1) * CWw + j;        // global token (may be < 0 -> zero pad)
    if (krow >= 0) {
      const _Float16* gk = Kh + ((long)(b * Tt + krow)) * Dd + h * Hd;
      const _Float16* gv = Vh + ((long)(b * Tt + krow)) * Dd + h * Hd;
#pragma unroll
      for (int c = 0; c < 64; c += 8) {
        *(v8h*)(sK + j * 64 + c) = *(const v8h*)(gk + c);
        v8h vv = *(const v8h*)(gv + c);
#pragma unroll
        for (int e = 0; e < 8; ++e) sVt[(c + e) * 256 + j] = vv[e];
      }
    } else {
      v8h z;
#pragma unroll
      for (int e = 0; e < 8; ++e) z[e] = (_Float16)0.0f;
#pragma unroll
      for (int c = 0; c < 64; c += 8) {
        *(v8h*)(sK + j * 64 + c) = z;
#pragma unroll
        for (int e = 0; e < 8; ++e) sVt[(c + e) * 256 + j] = (_Float16)0.0f;
      }
    }
  }
  __syncthreads();

  // ---- scores S[16 rows][256 cols] per wave; 16 accumulator tiles ----
  v8f S[16];
  const v16h aq0 = load_a_frag(sQ + (w * 16) * 64, 64);
  const v16h aq1 = load_a_frag(sQ + (w * 16) * 64 + 32, 64);
#pragma unroll
  for (int t = 0; t < 16; ++t) {
    v8f s;
#pragma unroll
    for (int v = 0; v < 8; ++v) s[v] = 0.0f;
    v16h bk0 = load_b_frag(sK + (t * 16) * 64, 64);      // B = K^T, K rows are K-contig
    v16h bk1 = load_b_frag(sK + (t * 16) * 64 + 32, 64);
    s = WMMA_F16(aq0, bk0, s);
    s = WMMA_F16(aq1, bk1, s);
    S[t] = s;
  }

  // ---- mask + scale + row softmax (rows live in one 16-lane half) ----
  const int   cn    = lane & 15;
  const int   hf    = lane >> 4;
  const float scale = 0.125f;                   // 1/sqrt(64)
#pragma unroll
  for (int v = 0; v < 8; ++v) {
    const int q = w * 16 + v + 8 * hf;          // block-local query index
    float m = -3.0e38f;
#pragma unroll
    for (int t = 0; t < 16; ++t) {
      const int j = t * 16 + cn;                // slab key index [0,256)
      const bool keep = (j >= q + 1) && (j <= q + CWw) && ((nb != 0) || (j <= 2 * q + 1));
      const float sv = keep ? S[t][v] * scale : -3.0e38f;
      S[t][v] = sv;
      m = fmaxf(m, sv);
    }
#pragma unroll
    for (int off = 1; off < 16; off <<= 1) m = fmaxf(m, __shfl_xor(m, off, 32));
    float sum = 0.0f;
#pragma unroll
    for (int t = 0; t < 16; ++t) {
      const float e = __expf(S[t][v] - m);
      S[t][v] = e;
      sum += e;
    }
#pragma unroll
    for (int off = 1; off < 16; off <<= 1) sum += __shfl_xor(sum, off, 32);
    const float inv = 1.0f / sum;               // window always has >=1 live element
#pragma unroll
    for (int t = 0; t < 16; ++t) S[t][v] *= inv;
  }

  // ---- O = P @ V  (two 128-column halves of P via LDS re-layout) ----
  v8f O[4];
#pragma unroll
  for (int j = 0; j < 4; ++j)
#pragma unroll
    for (int v = 0; v < 8; ++v) O[j][v] = 0.0f;

  for (int hk = 0; hk < 2; ++hk) {
    __syncthreads();                            // Q/K region dead, safe to overlay P
#pragma unroll
    for (int tt = 0; tt < 8; ++tt) {
      const int t = hk * 8 + tt;
#pragma unroll
      for (int v = 0; v < 8; ++v) {
        const int row = w * 16 + v + 8 * hf;
        sP[row * 128 + tt * 16 + cn] = (_Float16)S[t][v];
      }
    }
    __syncthreads();
#pragma unroll
    for (int ks = 0; ks < 4; ++ks) {
      v16h ap = load_a_frag(sP + (w * 16) * 128 + ks * 32, 128);
#pragma unroll
      for (int j = 0; j < 4; ++j) {
        v16h bv = load_b_frag(sVt + (j * 16) * 256 + hk * 128 + ks * 32, 256);
        O[j] = WMMA_F16(ap, bv, O[j]);
      }
    }
  }

  // ---- write attention output (f16, feeds projection GEMM) ----
#pragma unroll
  for (int j = 0; j < 4; ++j)
#pragma unroll
    for (int v = 0; v < 8; ++v) {
      const long row = (long)b * Tt + nb * CWw + w * 16 + v + 8 * hf;
      const long col = h * Hd + j * 16 + cn;
      AOh[row * Dd + col] = (_Float16)O[j][v];
    }
}

// ---------------------------------------------------------------------------
// Host-side launcher
// ---------------------------------------------------------------------------
extern "C" void kernel_launch(void* const* d_in, const int* in_sizes, int n_in,
                              void* d_out, int out_size, void* d_ws, size_t ws_size,
                              hipStream_t stream) {
  (void)in_sizes; (void)n_in; (void)out_size; (void)ws_size;

  const float* X  = (const float*)d_in[0];
  const float* Wq = (const float*)d_in[1];
  const float* Wk = (const float*)d_in[2];
  const float* Wv = (const float*)d_in[3];
  const float* Wp = (const float*)d_in[4];
  const float* bp = (const float*)d_in[5];
  float* out = (float*)d_out;

  // workspace layout (f16), all 16B aligned
  char* ws = (char*)d_ws;
  const size_t SZ_XD = (size_t)Mrows * Dd * sizeof(_Float16);   // 16 MB
  const size_t SZ_W  = (size_t)Dd * Dd * sizeof(_Float16);      //  2 MB
  _Float16* Xh  = (_Float16*)(ws);
  _Float16* Qh  = (_Float16*)(ws + SZ_XD);
  _Float16* Kh  = (_Float16*)(ws + 2 * SZ_XD);
  _Float16* Vh  = (_Float16*)(ws + 3 * SZ_XD);
  _Float16* AOh = (_Float16*)(ws + 4 * SZ_XD);
  _Float16* Wqt = (_Float16*)(ws + 5 * SZ_XD);                  // transposed [N][K]
  _Float16* Wkt = (_Float16*)(ws + 5 * SZ_XD + SZ_W);
  _Float16* Wvt = (_Float16*)(ws + 5 * SZ_XD + 2 * SZ_W);
  _Float16* Wpt = (_Float16*)(ws + 5 * SZ_XD + 3 * SZ_W);

  // 1) fp32 -> f16 conversions (weights transposed to K-major for TDM staging)
  const int nX = Mrows * Dd;     // 8388608
  f32_to_f16_kernel<<<nX / 1024, 256, 0, stream>>>(X, Xh, nX);
  dim3 tgrid(Dd / 32, Dd / 32);  // (32, 32)
  dim3 tblk(32, 8);
  f32_to_f16_transpose_kernel<<<tgrid, tblk, 0, stream>>>(Wq, Wqt, Dd, Dd);
  f32_to_f16_transpose_kernel<<<tgrid, tblk, 0, stream>>>(Wk, Wkt, Dd, Dd);
  f32_to_f16_transpose_kernel<<<tgrid, tblk, 0, stream>>>(Wv, Wvt, Dd, Dd);
  f32_to_f16_transpose_kernel<<<tgrid, tblk, 0, stream>>>(Wp, Wpt, Dd, Dd);

  // 2) QKV projections (f16 out, stays L2-resident), TDM-staged tiles
  dim3 ggrid(Dd / 128, Mrows / 128);   // (8, 64)
  gemm128_wmma_tdm<true><<<ggrid, 256, 0, stream>>>(Xh, Wqt, Qh, nullptr, nullptr, Mrows, Dd, Dd);
  gemm128_wmma_tdm<true><<<ggrid, 256, 0, stream>>>(Xh, Wkt, Kh, nullptr, nullptr, Mrows, Dd, Dd);
  gemm128_wmma_tdm<true><<<ggrid, 256, 0, stream>>>(Xh, Wvt, Vh, nullptr, nullptr, Mrows, Dd, Dd);

  // 3) sliding-window attention: 1024 blocks, 80KB dynamic LDS
  sliding_attn_wmma<<<dim3(Bc * NBb * NHh), 256, 81920, stream>>>(Qh, Kh, Vh, AOh);

  // 4) output projection (fp32 out + bias), TDM-staged tiles
  gemm128_wmma_tdm<false><<<ggrid, 256, 0, stream>>>(AOh, Wpt, nullptr, out, bp, Mrows, Dd, Dd);
}
